// SimpleAttention_5282809774629
// MI455X (gfx1250) — matrix operand order repaired and checked by reading, not compiled, not run on hardware
//
#include <hip/hip_runtime.h>

#define DIM 1024
#define QKV_N 3072
#define HEADS 16
#define HEAD_DIM 64

// GEMM blocking: 128x128 block tile, K staged in 64-wide double-buffered LDS panels.
#define TM 128
#define TN 128
#define TK 64

#if __has_builtin(__builtin_amdgcn_global_load_async_to_lds_b128)
#define USE_ASYNC_B 1
#else
#define USE_ASYNC_B 0
#endif

typedef __attribute__((ext_vector_type(16))) __bf16 v16bf;
typedef __attribute__((ext_vector_type(8)))  __bf16 v8bf;
typedef __attribute__((ext_vector_type(4)))  __bf16 v4bf;
typedef __attribute__((ext_vector_type(8)))  float  v8f;
typedef __attribute__((ext_vector_type(4)))  float  v4f;
typedef __attribute__((ext_vector_type(4)))  int    v4i;

// address-space-qualified pointer types for the async global->LDS builtin
// (ROCm 7.2 signature: (v4i AS1* gsrc, v4i AS3* lds, Ii offset, Ii cpol))
typedef v4i __attribute__((address_space(1))) *as1_v4i_ptr;
typedef v4i __attribute__((address_space(3))) *as3_v4i_ptr;

// hardware bf16 convert (RNE) -- let clang emit v_cvt instead of bit twiddling
__device__ __forceinline__ __bf16 f2bf(float f) { return (__bf16)f; }

// ---------------- one-time weight fp32 -> bf16 (8MB, stays L2-resident) ----------------
__global__ void wconvert_kernel(const float* __restrict__ wq, const float* __restrict__ wp,
                                __bf16* __restrict__ wqb, __bf16* __restrict__ wpb,
                                int nq, int np) {
    int i = blockIdx.x * blockDim.x + threadIdx.x;
    int stride = gridDim.x * blockDim.x;
    for (int k = i; k < nq; k += stride) wqb[k] = f2bf(wq[k]);
    for (int k = i; k < np; k += stride) wpb[k] = f2bf(wp[k]);
}

// ---------------- D = A @ B^T + bias  (A:[Mtot x K], B:[Ntot x K] bf16) ----------------
// Block: 128(M) x 128(N); 8 waves in a 4x2 grid; wave tile 32x64 = 2x4 WMMA accumulators.
template <bool A_F32, bool D_BF16>
__global__ __launch_bounds__(256)
void gemm_bias_kernel(const void* __restrict__ Ap, const __bf16* __restrict__ Bp,
                      const float* __restrict__ bias, void* __restrict__ Dp,
                      int K, int Ntot) {
    extern __shared__ char smem[];
    __bf16* As = (__bf16*)smem;                               // 2 x (128 x 64) bf16
    __bf16* Bs = (__bf16*)(smem + 2 * TM * TK * sizeof(__bf16));

    const int tid = threadIdx.x, wave = tid >> 5, lane = tid & 31;
    const int half = lane >> 4, lr = lane & 15;
    const int wm = wave >> 1, wn = wave & 1;                  // 4x2 wave grid
    const int Mb = blockIdx.y * TM, Nb = blockIdx.x * TN;

    // accumulators initialized with per-column bias (C/D layout: col N = lr)
    v8f acc[2][4];
    #pragma unroll
    for (int nt = 0; nt < 4; ++nt) {
        float bv = bias[Nb + wn * 64 + nt * 16 + lr];
        #pragma unroll
        for (int mt = 0; mt < 2; ++mt)
            #pragma unroll
            for (int v = 0; v < 8; ++v) acc[mt][nt][v] = bv;
    }

    const int P = K / TK;

    v4f  areg_f[8];   // A staging regs (fp32 source path)
    v8bf areg_h[4];   // A staging regs (bf16 source path)
    v8bf breg[4];     // B staging regs (sync fallback path)

    auto loadA = [&](int p) {
        const int kb = p * TK;
        if constexpr (A_F32) {
            const float* A = (const float*)Ap;
            #pragma unroll
            for (int i = 0; i < 8; ++i) {
                int flat = (tid + i * 256) * 4, r = flat >> 6, c = flat & 63;
                areg_f[i] = *(const v4f*)(A + (size_t)(Mb + r) * K + kb + c);
            }
        } else {
            const __bf16* A = (const __bf16*)Ap;
            #pragma unroll
            for (int i = 0; i < 4; ++i) {
                int flat = (tid + i * 256) * 8, r = flat >> 6, c = flat & 63;
                areg_h[i] = *(const v8bf*)(A + (size_t)(Mb + r) * K + kb + c);
            }
        }
    };
    auto storeA = [&](int buf) {
        __bf16* Ad = As + buf * TM * TK;
        if constexpr (A_F32) {
            #pragma unroll
            for (int i = 0; i < 8; ++i) {
                int flat = (tid + i * 256) * 4, r = flat >> 6, c = flat & 63;
                v4bf pk;
                pk[0] = f2bf(areg_f[i][0]); pk[1] = f2bf(areg_f[i][1]);
                pk[2] = f2bf(areg_f[i][2]); pk[3] = f2bf(areg_f[i][3]);
                *(v4bf*)(Ad + r * TK + c) = pk;
            }
        } else {
            #pragma unroll
            for (int i = 0; i < 4; ++i) {
                int flat = (tid + i * 256) * 8, r = flat >> 6, c = flat & 63;
                *(v8bf*)(Ad + r * TK + c) = areg_h[i];
            }
        }
    };
    auto loadB = [&](int p) {
        const int kb = p * TK;
        #pragma unroll
        for (int i = 0; i < 4; ++i) {
            int flat = (tid + i * 256) * 8, r = flat >> 6, c = flat & 63;
            breg[i] = *(const v8bf*)(Bp + (size_t)(Nb + r) * K + kb + c);
        }
    };
    auto storeB = [&](int buf) {
        __bf16* Bd = Bs + buf * TM * TK;
        #pragma unroll
        for (int i = 0; i < 4; ++i) {
            int flat = (tid + i * 256) * 8, r = flat >> 6, c = flat & 63;
            *(v8bf*)(Bd + r * TK + c) = breg[i];
        }
    };
#if USE_ASYNC_B
    // B panel: per-lane 16B global->LDS async copies, tracked by ASYNCcnt.
    auto asyncB = [&](int p, int buf) {
        const int kb = p * TK;
        __bf16* Bd = Bs + buf * TM * TK;
        #pragma unroll
        for (int i = 0; i < 4; ++i) {
            int flat = (tid + i * 256) * 8, r = flat >> 6, c = flat & 63;
            const __bf16* src = Bp + (size_t)(Nb + r) * K + kb + c;
            __bf16* dst = Bd + r * TK + c;
            __builtin_amdgcn_global_load_async_to_lds_b128(
                (as1_v4i_ptr)(uintptr_t)src,
                (as3_v4i_ptr)(uintptr_t)dst, 0, 0);
        }
    };
#endif
    auto compute = [&](int buf) {
        const __bf16* Ab = As + buf * TM * TK;
        const __bf16* Bb = Bs + buf * TM * TK;
        #pragma unroll
        for (int kk = 0; kk < TK; kk += 32) {
            v16bf afr[2], bfr[4];
            #pragma unroll
            for (int mt = 0; mt < 2; ++mt) {       // A frag: ISA 7.12.2 16-bit A layout
                const __bf16* ar = Ab + (wm * 32 + mt * 16 + lr) * TK;
                v8bf lo = *(const v8bf*)(ar + kk + 8 * half);
                v8bf hi = *(const v8bf*)(ar + kk + 16 + 8 * half);
                afr[mt] = __builtin_shufflevector(lo, hi,
                    0, 1, 2, 3, 4, 5, 6, 7, 8, 9, 10, 11, 12, 13, 14, 15);
            }
            #pragma unroll
            for (int nt = 0; nt < 4; ++nt) {        // B frag: same k order as A (8h+s, 16+8h+s-8)
                const __bf16* br = Bb + (wn * 64 + nt * 16 + lr) * TK;
                v8bf blo = *(const v8bf*)(br + kk + 8 * half);
                v8bf bhi = *(const v8bf*)(br + kk + 16 + 8 * half);
                bfr[nt] = __builtin_shufflevector(blo, bhi,
                    0, 1, 2, 3, 4, 5, 6, 7, 8, 9, 10, 11, 12, 13, 14, 15);
            }
            #pragma unroll
            for (int mt = 0; mt < 2; ++mt)
                #pragma unroll
                for (int nt = 0; nt < 4; ++nt)      // 8 WMMAs per k-step: 2x A, 4x B reuse
                    acc[mt][nt] = __builtin_amdgcn_wmma_f32_16x16x32_bf16(
                        false, afr[mt], false, bfr[nt], (short)0, acc[mt][nt], false, false);
        }
    };

    // software-pipelined main loop: panel p+1 transfers in flight during compute(p)
#if USE_ASYNC_B
    asyncB(0, 0); loadA(0);
    for (int p = 0; p < P; ++p) {
        int buf = p & 1;
        storeA(buf);
        asm volatile("s_wait_asynccnt 0x0" ::: "memory");  // panel p B copies landed in LDS
        __syncthreads();
        if (p + 1 < P) { asyncB(p + 1, buf ^ 1); loadA(p + 1); }
        compute(buf);
        __syncthreads();
    }
#else
    loadA(0); loadB(0);
    for (int p = 0; p < P; ++p) {
        int buf = p & 1;
        storeA(buf); storeB(buf);
        __syncthreads();
        if (p + 1 < P) { loadA(p + 1); loadB(p + 1); }
        compute(buf);
        __syncthreads();
    }
#endif

    // epilogue: D layout row M = v + 8*half, col N = lr
    #pragma unroll
    for (int mt = 0; mt < 2; ++mt)
        #pragma unroll
        for (int nt = 0; nt < 4; ++nt) {
            int col = Nb + wn * 64 + nt * 16 + lr;
            #pragma unroll
            for (int v = 0; v < 8; ++v) {
                int row = Mb + wm * 32 + mt * 16 + v + 8 * half;
                if constexpr (D_BF16)
                    ((__bf16*)Dp)[(size_t)row * Ntot + col] = f2bf(acc[mt][nt][v]);
                else
                    ((float*)Dp)[(size_t)row * Ntot + col] = acc[mt][nt][v];
            }
        }
}

// ---------------- per-token cross-head attention (16x16 softmax), ~0.8% of FLOPs ----------------
__global__ __launch_bounds__(256)
void attn_kernel(const __bf16* __restrict__ qkv, __bf16* __restrict__ aout) {
    extern __shared__ char smem[];
    __bf16* ldsQ = (__bf16*)smem;                   // 16 x 3072 bf16 (96KB)
    float*  ldsS = (float*)(smem + 16 * QKV_N * 2); // 16 x 256 f32  (16KB)
    const int tid = threadIdx.x;
    const int m0 = blockIdx.x * 16;

    for (int idx = tid; idx < 16 * (QKV_N / 8); idx += 256) {
        int r = idx / (QKV_N / 8), c = (idx - r * (QKV_N / 8)) * 8;
        *(v8bf*)(ldsQ + r * QKV_N + c) = *(const v8bf*)(qkv + (size_t)(m0 + r) * QKV_N + c);
    }
    __syncthreads();
    for (int s = tid; s < 16 * 256; s += 256) {     // scores[t][h][g] = q.k / 8
        int t = s >> 8, h = (s >> 4) & 15, g = s & 15;
        const __bf16* qp = ldsQ + t * QKV_N + h * HEAD_DIM;
        const __bf16* kp = ldsQ + t * QKV_N + DIM + g * HEAD_DIM;
        float d = 0.f;
        #pragma unroll 8
        for (int e = 0; e < HEAD_DIM; ++e) d += (float)qp[e] * (float)kp[e];
        ldsS[s] = d * 0.125f;
    }
    __syncthreads();
    {   // softmax over g: one (t,h) row per thread
        float* row = ldsS + tid * 16;
        float mx = row[0];
        #pragma unroll
        for (int g = 1; g < 16; ++g) mx = fmaxf(mx, row[g]);
        float sum = 0.f;
        #pragma unroll
        for (int g = 0; g < 16; ++g) { float e = __expf(row[g] - mx); row[g] = e; sum += e; }
        float inv = 1.f / sum;
        #pragma unroll
        for (int g = 0; g < 16; ++g) row[g] *= inv;
    }
    __syncthreads();
    for (int f = tid; f < 16 * DIM; f += 256) {     // out[t][h][d] = attn @ v
        int t = f >> 10, c = f & (DIM - 1), h = c >> 6, d = c & 63;
        const float* ar = ldsS + t * 256 + h * 16;
        const __bf16* vp = ldsQ + t * QKV_N + 2 * DIM + d;
        float s = 0.f;
        #pragma unroll
        for (int g = 0; g < 16; ++g) s += ar[g] * (float)vp[g * HEAD_DIM];
        aout[(size_t)(m0 + t) * DIM + c] = f2bf(s);
    }
}

extern "C" void kernel_launch(void* const* d_in, const int* in_sizes, int n_in,
                              void* d_out, int out_size, void* d_ws, size_t ws_size,
                              hipStream_t stream) {
    const float* x  = (const float*)d_in[0];
    const float* wq = (const float*)d_in[1];
    const float* bq = (const float*)d_in[2];
    const float* wp = (const float*)d_in[3];
    const float* bp = (const float*)d_in[4];
    float* out = (float*)d_out;

    const int nq = in_sizes[1];                 // 3*DIM*DIM
    const int np = in_sizes[3];                 // DIM*DIM
    const int ntok = in_sizes[0] / DIM;         // B*N = 16384

    // workspace layout (total ~136MB; qkv+attn intermediates sized to live in 192MB L2)
    size_t off = 0;
    __bf16* wqb  = (__bf16*)((char*)d_ws + off); off += (size_t)nq * 2;
    __bf16* wpb  = (__bf16*)((char*)d_ws + off); off += (size_t)np * 2;
    __bf16* qkvb = (__bf16*)((char*)d_ws + off); off += (size_t)ntok * QKV_N * 2;
    __bf16* aob  = (__bf16*)((char*)d_ws + off);

    wconvert_kernel<<<1024, 256, 0, stream>>>(wq, wp, wqb, wpb, nq, np);

    const size_t glds = (size_t)4 * TM * TK * sizeof(__bf16);  // 64KB (2 bufs x A,B)
    const size_t alds = (size_t)16 * QKV_N * 2 + 16 * 256 * 4; // 112KB
    (void)hipFuncSetAttribute(reinterpret_cast<const void*>(&gemm_bias_kernel<true, true>),
                              hipFuncAttributeMaxDynamicSharedMemorySize, (int)glds);
    (void)hipFuncSetAttribute(reinterpret_cast<const void*>(&gemm_bias_kernel<false, false>),
                              hipFuncAttributeMaxDynamicSharedMemorySize, (int)glds);
    (void)hipFuncSetAttribute(reinterpret_cast<const void*>(&attn_kernel),
                              hipFuncAttributeMaxDynamicSharedMemorySize, (int)alds);

    // qkv = x @ Wqkv^T + b   -> bf16 ws
    gemm_bias_kernel<true, true>
        <<<dim3(QKV_N / TN, ntok / TM), 256, glds, stream>>>(x, wqb, bq, qkvb, DIM, QKV_N);
    // per-token head attention -> bf16 ws
    attn_kernel<<<ntok / 16, 256, alds, stream>>>(qkvb, aob);
    // out = attn_out @ Wproj^T + b -> fp32
    gemm_bias_kernel<false, false>
        <<<dim3(DIM / TN, ntok / TM), 256, glds, stream>>>(aob, wpb, bp, out, DIM, DIM);
}
